// BeatFCOS_90082644066646
// MI455X (gfx1250) — compile-verified, matrix operand order checked by reading, not run on hardware
//
#include <hip/hip_runtime.h>
#include <hip/hip_bf16.h>
#include <math.h>

typedef __attribute__((ext_vector_type(16))) __bf16 v16bf;
typedef __attribute__((ext_vector_type(8)))  float  v8f;

#define FEAT 256
#define NB 8
#define LTOT 15872

__device__ __forceinline__ unsigned short f2bf(float f) {
    unsigned u = __float_as_uint(f);
    unsigned r = u + 0x7FFFu + ((u >> 16) & 1u);   // round-to-nearest-even
    return (unsigned short)(r >> 16);
}
__device__ __forceinline__ float bf2f(unsigned short h) {
    return __uint_as_float(((unsigned)h) << 16);
}

// ---------------------------------------------------------------------------
// Weight pre-pack: fp32 [256, Cin, taps] -> bf16 fragments in the exact
// per-lane layout of the WMMA 16-bit A matrix (16x32 per tile).
// Linear order: chunk(tap,cc32) -> ctile(16) -> lane(32) -> elem(16).
// Each lane's fragment is then 32 contiguous bytes (2x global_load_b128).
// ---------------------------------------------------------------------------
__global__ void pack_w(const float* __restrict__ W, unsigned short* __restrict__ out,
                       int Cin, int taps, int total)
{
    int j = blockIdx.x * 256 + threadIdx.x;
    if (j >= total) return;
    int e     = j & 15;
    int t     = (j >> 4) & 31;
    int ctile = (j >> 9) & 15;
    int chunk = j >> 13;
    int nCC = Cin >> 5;
    int tap = chunk / nCC, ccI = chunk - tap * nCC;
    int m = t & 15, kb = (t >> 4) * 8;
    int K = (e < 8) ? (kb + e) : (kb + 8 + e);
    int ci = ccI * 32 + K;
    int co = ctile * 16 + m;
    out[j] = f2bf(W[((size_t)co * Cin + ci) * taps + tap]);
}

// ---------------------------------------------------------------------------
// conv1d (Cout=256) as implicit GEMM on v_wmma_f32_16x16x32_bf16.
// Block = 256 threads (8 waves), produces [256 Cout] x [64/STRIDE positions].
// LDS tile layout: [position][channel] (row = 264 bf16, 16B-aligned rows),
// so each B fragment is two ds_load_b128. A fragments are two
// global_load_b128 from the pre-packed bf16 weights (L2-resident, reused
// across all blocks and all NT position tiles).
// ---------------------------------------------------------------------------
template<int TAPS, int STRIDE>
__launch_bounds__(256)
__global__ void conv256_wmma(const float* __restrict__ X,
                             const uint4* __restrict__ Wq,   // packed bf16 fragments
                             const float* __restrict__ Bias,
                             float* __restrict__ Y,
                             const float* __restrict__ Res,  // optional, index l>>1
                             int Lres, int Cin, int Lin, int Lout, int inRelu)
{
    constexpr int POS  = 64 / STRIDE;           // output positions per block
    constexpr int NT   = POS / 16;              // position tiles per wave
    constexpr int P    = (POS - 1) * STRIDE + TAPS;  // input positions needed
    constexpr int PAD  = TAPS / 2;
    constexpr int ROWE = 264;                   // padded row (132 dwords = 4 mod 64 banks)
    __shared__ unsigned short xs[66 * ROWE];

    const int b  = blockIdx.y;
    const int l0 = blockIdx.x * POS;
    const int p0 = l0 * STRIDE - PAD;

    // Stage input tile fp32 -> bf16, transposed to [p][ci], zero-padded edges.
    for (int i = threadIdx.x; i < Cin * P; i += 256) {
        int ci = i / P, p = i - ci * P;
        int gp = p0 + p;
        float v = 0.0f;
        if ((unsigned)gp < (unsigned)Lin)
            v = X[((size_t)(b * Cin + ci)) * Lin + gp];
        if (inRelu) v = fmaxf(v, 0.0f);
        xs[p * ROWE + ci] = f2bf(v);
    }
    __syncthreads();

    const int w  = threadIdx.x >> 5;
    const int t  = threadIdx.x & 31;
    const int m  = t & 15;
    const int hi = t >> 4;
    const int kb = hi * 8;
    const int ctile0 = w * 2;    // wave owns Cout [32w, 32w+32)

    v8f zero = {0,0,0,0,0,0,0,0};
    v8f acc[2][NT];
    #pragma unroll
    for (int ct = 0; ct < 2; ++ct)
        #pragma unroll
        for (int pt = 0; pt < NT; ++pt) acc[ct][pt] = zero;

    union U { v16bf v; uint4 q[2]; };
    const int nCC = Cin >> 5;

    #pragma unroll
    for (int tap = 0; tap < TAPS; ++tap) {
        for (int ccI = 0; ccI < nCC; ++ccI) {
            const int chunk = tap * nCC + ccI;
            const uint4* ap = Wq + ((size_t)(chunk * 16 + ctile0)) * 64 + t * 2;
            U A0, A1;
            A0.q[0] = ap[0];   A0.q[1] = ap[1];    // ctile0
            A1.q[0] = ap[64];  A1.q[1] = ap[65];   // ctile0+1
            const int cbase = ccI * 32 + kb;
            #pragma unroll
            for (int pt = 0; pt < NT; ++pt) {
                const int p = (pt * 16 + m) * STRIDE + tap;
                const uint4* bp = (const uint4*)&xs[p * ROWE + cbase];
                U Bf;
                Bf.q[0] = bp[0];    // K = kb + 0..7
                Bf.q[1] = bp[2];    // K = kb + 16..23
                acc[0][pt] = __builtin_amdgcn_wmma_f32_16x16x32_bf16(
                    false, A0.v, false, Bf.v, (short)0, acc[0][pt], false, false);
                acc[1][pt] = __builtin_amdgcn_wmma_f32_16x16x32_bf16(
                    false, A1.v, false, Bf.v, (short)0, acc[1][pt], false, false);
            }
        }
    }

    // D layout: lane -> N = t&15, VGPR r -> M = r + (t>>4)*8
    const int hi8 = hi * 8;
    #pragma unroll
    for (int ct = 0; ct < 2; ++ct) {
        const int coB = (ctile0 + ct) * 16 + hi8;
        #pragma unroll
        for (int pt = 0; pt < NT; ++pt) {
            const int l = l0 + pt * 16 + m;
            if (l >= Lout) continue;
            #pragma unroll
            for (int r = 0; r < 8; ++r) {
                const int co = coB + r;
                const size_t row = (size_t)(b * FEAT + co);
                float v = acc[ct][pt][r] + Bias[co];
                if (Res) v += Res[row * Lres + (l >> 1)];   // upsample2-add fusion
                Y[row * Lout + l] = v;
            }
        }
    }
}

// ---------------------------------------------------------------------------
// GroupNorm (32 groups x 8 ch): stats per (b,group), then in-place affine+ReLU.
// ---------------------------------------------------------------------------
__global__ void gn_stats(const float* __restrict__ X, float* __restrict__ stats, int L)
{
    int g = blockIdx.x, b = blockIdx.y;
    const float* base = X + ((size_t)(b * FEAT) + g * 8) * L;
    int N = 8 * L;
    float s = 0.f, s2 = 0.f;
    for (int i = threadIdx.x; i < N; i += 256) {
        float v = base[i];
        s += v; s2 += v * v;
    }
    __shared__ float sh[512];
    sh[threadIdx.x] = s; sh[256 + threadIdx.x] = s2;
    __syncthreads();
    for (int o = 128; o > 0; o >>= 1) {
        if (threadIdx.x < o) {
            sh[threadIdx.x]       += sh[threadIdx.x + o];
            sh[256 + threadIdx.x] += sh[256 + threadIdx.x + o];
        }
        __syncthreads();
    }
    if (threadIdx.x == 0) {
        float mean = sh[0] / N;
        float var  = sh[256] / N - mean * mean;
        stats[(b * 32 + g) * 2 + 0] = mean;
        stats[(b * 32 + g) * 2 + 1] = rsqrtf(var + 1e-5f);
    }
}

__global__ void gn_apply(float* __restrict__ X, const float* __restrict__ stats,
                         const float* __restrict__ gamma, const float* __restrict__ beta, int L)
{
    int b = blockIdx.y;
    long long idx = (long long)blockIdx.x * 256 + threadIdx.x;
    if (idx >= (long long)FEAT * L) return;
    int c = (int)(idx / L);
    int g = c >> 3;
    float m  = stats[(b * 32 + g) * 2 + 0];
    float rs = stats[(b * 32 + g) * 2 + 1];
    float* p = X + (size_t)b * FEAT * L + idx;
    float v = (*p - m) * rs * gamma[c] + beta[c];
    *p = fmaxf(v, 0.0f);
}

// ---------------------------------------------------------------------------
// Tiny output convs (Cout 1..2, k=3) + optional sigmoid; writes transposed
// into the concatenated [B, LTOT, Co] output section.
// ---------------------------------------------------------------------------
__global__ void head_out(const float* __restrict__ X,     // [B,256,L]
                         const float* __restrict__ W,     // [Co,256,3]
                         const float* __restrict__ Bias,
                         float* __restrict__ out,
                         int Co, int sig, int L, int lvlOff)
{
    __shared__ unsigned short xs[FEAT * 66];
    __shared__ float ww[2 * FEAT * 3];

    int b  = blockIdx.y;
    int l0 = blockIdx.x * 64;

    for (int i = threadIdx.x; i < FEAT * 66; i += 64) {
        int ci = i / 66, p = i - ci * 66;
        int gp = l0 - 1 + p;
        float v = ((unsigned)gp < (unsigned)L)
                    ? X[((size_t)(b * FEAT + ci)) * L + gp] : 0.0f;
        xs[ci * 66 + p] = f2bf(v);
    }
    for (int i = threadIdx.x; i < Co * FEAT * 3; i += 64) ww[i] = W[i];
    __syncthreads();

    int l = l0 + threadIdx.x;
    if (l >= L) return;
    float acc0 = Bias[0];
    float acc1 = (Co > 1) ? Bias[1] : 0.0f;
    for (int ci = 0; ci < FEAT; ++ci) {
        float x0 = bf2f(xs[ci * 66 + threadIdx.x + 0]);
        float x1 = bf2f(xs[ci * 66 + threadIdx.x + 1]);
        float x2 = bf2f(xs[ci * 66 + threadIdx.x + 2]);
        const float* w0 = &ww[ci * 3];
        acc0 += x0 * w0[0] + x1 * w0[1] + x2 * w0[2];
        if (Co > 1) {
            const float* w1 = &ww[(FEAT + ci) * 3];
            acc1 += x0 * w1[0] + x1 * w1[1] + x2 * w1[2];
        }
    }
    size_t pos = (size_t)b * LTOT + lvlOff + l;
    if (sig) acc0 = 1.0f / (1.0f + expf(-acc0));
    out[pos * Co + 0] = acc0;
    if (Co > 1) {
        if (sig) acc1 = 1.0f / (1.0f + expf(-acc1));
        out[pos * Co + 1] = acc1;
    }
}

// ---------------------------------------------------------------------------
extern "C" void kernel_launch(void* const* d_in, const int* in_sizes, int n_in,
                              void* d_out, int out_size, void* d_ws, size_t ws_size,
                              hipStream_t stream)
{
    (void)in_sizes; (void)n_in; (void)out_size; (void)ws_size;
    const float* C4 = (const float*)d_in[0];   // [8,128,8192]
    const float* C5 = (const float*)d_in[1];   // [8,256,4096]
    auto f = [&](int i) { return (const float*)d_in[i]; };

    // Workspace layout (floats)
    float* ws   = (float*)d_ws;
    float* fmP4 = ws;
    float* fmP5 = fmP4 + (size_t)NB * FEAT * 8192;
    float* fmP6 = fmP5 + (size_t)NB * FEAT * 4096;
    float* fmP7 = fmP6 + (size_t)NB * FEAT * 2048;
    float* fmP8 = fmP7 + (size_t)NB * FEAT * 1024;
    float* tmpA = fmP8 + (size_t)NB * FEAT * 512;
    float* tmpB = tmpA + (size_t)NB * FEAT * 8192;
    float* stats = tmpB + (size_t)NB * FEAT * 8192;          // 512 floats
    unsigned short* wpkBase = (unsigned short*)(stats + 512); // packed bf16 weights

    // Pre-pack all GEMM weights (bf16, WMMA A-fragment order).
    size_t wOff = 0;
    auto pack = [&](int idx, int Cin, int taps) -> const unsigned short* {
        unsigned short* dst = wpkBase + wOff;
        int total = FEAT * Cin * taps;
        hipLaunchKernelGGL(pack_w, dim3((total + 255) / 256), dim3(256), 0, stream,
                           f(idx), dst, Cin, taps, total);
        wOff += (size_t)total;
        return dst;
    };
    const unsigned short* pw51 = pack(2, 256, 1);
    const unsigned short* pw52 = pack(4, 256, 3);
    const unsigned short* pw41 = pack(6, 128, 1);
    const unsigned short* pw42 = pack(8, 256, 3);
    const unsigned short* pw6  = pack(10, 256, 3);
    const unsigned short* pw7  = pack(12, 256, 3);
    const unsigned short* pw8  = pack(14, 256, 3);
    const unsigned short* pwc1 = pack(16, 256, 3);
    const unsigned short* pwc2 = pack(20, 256, 3);
    const unsigned short* pwr1 = pack(26, 256, 3);
    const unsigned short* pwr2 = pack(30, 256, 3);

    auto conv = [&](const float* X, const unsigned short* Wp, const float* Bb, float* Y,
                    const float* Res, int Lres, int Cin, int Lin, int Lout,
                    int taps, int stride, int inRelu) {
        const uint4* Wq = (const uint4*)Wp;
        if (taps == 1 && stride == 1) {
            hipLaunchKernelGGL((conv256_wmma<1,1>), dim3((Lout + 63) / 64, NB), dim3(256), 0,
                               stream, X, Wq, Bb, Y, Res, Lres, Cin, Lin, Lout, inRelu);
        } else if (taps == 3 && stride == 1) {
            hipLaunchKernelGGL((conv256_wmma<3,1>), dim3((Lout + 63) / 64, NB), dim3(256), 0,
                               stream, X, Wq, Bb, Y, Res, Lres, Cin, Lin, Lout, inRelu);
        } else {
            hipLaunchKernelGGL((conv256_wmma<3,2>), dim3((Lout + 31) / 32, NB), dim3(256), 0,
                               stream, X, Wq, Bb, Y, Res, Lres, Cin, Lin, Lout, inRelu);
        }
    };
    auto gn = [&](float* Xp, const float* gamma, const float* beta, int L) {
        hipLaunchKernelGGL(gn_stats, dim3(32, NB), dim3(256), 0, stream, Xp, stats, L);
        int total = FEAT * L;
        hipLaunchKernelGGL(gn_apply, dim3((total + 255) / 256, NB), dim3(256), 0, stream,
                           Xp, stats, gamma, beta, L);
    };
    auto hout = [&](const float* T, const float* Wt, const float* Bb, float* sec,
                    int Co, int sig, int L, int off) {
        hipLaunchKernelGGL(head_out, dim3((L + 63) / 64, NB), dim3(64), 0, stream,
                           T, Wt, Bb, sec, Co, sig, L, off);
    };

    // ---- FPN ----
    conv(C5,   pw51, f(3),  tmpB, nullptr, 0,    256, 4096, 4096, 1, 1, 0); // P5a
    conv(tmpB, pw52, f(5),  fmP5, nullptr, 0,    256, 4096, 4096, 3, 1, 0); // P5
    conv(C4,   pw41, f(7),  tmpA, tmpB,    4096, 128, 8192, 8192, 1, 1, 0); // P4a (+up2(P5a))
    conv(tmpA, pw42, f(9),  fmP4, nullptr, 0,    256, 8192, 8192, 3, 1, 0); // P4
    conv(C5,   pw6,  f(11), fmP6, nullptr, 0,    256, 4096, 2048, 3, 2, 0); // P6
    conv(fmP6, pw7,  f(13), fmP7, nullptr, 0,    256, 2048, 1024, 3, 2, 1); // P7 (relu in)
    conv(fmP7, pw8,  f(15), fmP8, nullptr, 0,    256, 1024, 512,  3, 2, 1); // P8 (relu in)

    // ---- Heads ----
    float* fms[5]  = { fmP4, fmP5, fmP6, fmP7, fmP8 };
    int    Ls[5]   = { 8192, 4096, 2048, 1024, 512 };
    int    offs[5] = { 0, 8192, 12288, 14336, 15360 };

    float* outp   = (float*)d_out;
    float* clsSec = outp;
    float* regSec = outp + (size_t)NB * LTOT * 2;
    float* lftSec = outp + (size_t)NB * LTOT * 4;

    for (int i = 0; i < 5; ++i) {
        int L = Ls[i], off = offs[i];
        // cls head
        conv(fms[i], pwc1, f(17), tmpA, nullptr, 0, 256, L, L, 3, 1, 0);
        gn(tmpA, f(18), f(19), L);
        conv(tmpA,   pwc2, f(21), tmpB, nullptr, 0, 256, L, L, 3, 1, 0);
        gn(tmpB, f(22), f(23), L);
        hout(tmpB, f(24), f(25), clsSec, 2, 1, L, off);
        // reg head
        conv(fms[i], pwr1, f(27), tmpA, nullptr, 0, 256, L, L, 3, 1, 0);
        gn(tmpA, f(28), f(29), L);
        conv(tmpA,   pwr2, f(31), tmpB, nullptr, 0, 256, L, L, 3, 1, 0);
        gn(tmpB, f(32), f(33), L);
        hout(tmpB, f(34), f(35), regSec, 2, 0, L, off);
        hout(tmpB, f(36), f(37), lftSec, 1, 1, L, off);
    }
}
